// Attention_29867202576782
// MI455X (gfx1250) — matrix operand order repaired and checked by reading, not compiled, not run on hardware
//
#include <hip/hip_runtime.h>

typedef __attribute__((ext_vector_type(16))) _Float16 v16h;
typedef __attribute__((ext_vector_type(8)))  float    v8f;

#define B_    16
#define SQ_   32
#define STOT  2048
#define H_    32
#define HKV_  8
#define D_    128
#define G_    4
#define SCTX  (STOT - SQ_)          /* 2016 = 63*32: chunks never straddle */
#define SCALE_ 0.08838834764831845f
#define SKV   32                    /* kv positions per chunk   */
#define NCHUNK (STOT / SKV)         /* 64 chunks                */

// DPP16 ROW_XMASK: lane <- lane^M within each 16-lane row (pure VALU, no LDS)
template<int M>
__device__ __forceinline__ float dpp_xor(float x) {
    int r = __builtin_amdgcn_update_dpp(0, __builtin_bit_cast(int, x),
                                        0x160 | M, 0xF, 0xF, true);
    return __builtin_bit_cast(float, r);
}
__device__ __forceinline__ float row_max16(float x) {
    x = fmaxf(x, dpp_xor<1>(x));
    x = fmaxf(x, dpp_xor<2>(x));
    x = fmaxf(x, dpp_xor<4>(x));
    x = fmaxf(x, dpp_xor<8>(x));
    return x;
}
__device__ __forceinline__ float row_sum16(float x) {
    x += dpp_xor<1>(x);
    x += dpp_xor<2>(x);
    x += dpp_xor<4>(x);
    x += dpp_xor<8>(x);
    return x;
}

__global__ __launch_bounds__(256, 1)
void attn_gqa_wmma(const float* __restrict__ q,
                   const float* __restrict__ knew,
                   const float* __restrict__ vnew,
                   const float* __restrict__ kcache,
                   const float* __restrict__ vcache,
                   float* __restrict__ out)
{
    // double-buffered chunk tiles (+pad against bank conflicts)
    __shared__ _Float16 lds_k [2][SKV][D_ + 8];   // K chunk, [s][d]
    __shared__ _Float16 lds_vt[2][D_][SKV + 8];   // V chunk transposed, [d][s]
    __shared__ _Float16 lds_p [8][16][SKV];       // per-wave P staging (C->A relayout)

    const int tid  = threadIdx.x;
    const int w    = tid >> 5;          // wave 0..7
    const int lane = tid & 31;
    const int half = lane >> 4;         // lane half (A/B/C layout selector)
    const int l16  = lane & 15;

    const int b   = blockIdx.x / HKV_;
    const int hkv = blockIdx.x % HKV_;

    // ---------------- Q fragments: A-matrix 16x32 f16, 4 K-steps cover D=128 ----
    // A layout: M = lane%16 ; K = half*16 + j
    v16h qa[4];
    {
        const int qr = w * 16 + l16;                 // query row within (b,hkv)
        const int sq = qr >> 2, g = qr & 3;          // qr = sq*G + g
        const float* qrow = q + (size_t)(b * SQ_ + sq) * (H_ * D_)
                              + (size_t)(hkv * G_ + g) * D_;
#pragma unroll
        for (int kk = 0; kk < 4; ++kk)
#pragma unroll
            for (int j = 0; j < 16; ++j) {
                int kl = half * 16 + j;
                qa[kk][j] = (_Float16)(qrow[kk * 32 + kl] * SCALE_);
            }
    }

    // ---------------- running softmax state + O accumulators --------------------
    float m_run[8], l_run[8];
    v8f   o_acc[8];                                   // 16x128 f32 per wave
#pragma unroll
    for (int r = 0; r < 8; ++r) { m_run[r] = -3.0e38f; l_run[r] = 0.0f; }
#pragma unroll
    for (int nd = 0; nd < 8; ++nd) {
        v8f z = {};
        o_acc[nd] = z;
    }

    // register-staged prefetch of one K/V chunk (4 float4 per matrix per thread)
    float4 kreg[4], vreg[4];
    auto load_chunk = [&](int c) {
        const int sbase = c * SKV;
        // both sources stride 1024 floats per kv position; uniform base select
        const float* kp;
        const float* vp;
        if (sbase >= SCTX) {          // chunk 63: entirely new tokens
            size_t base = (size_t)(b * SQ_ + (sbase - SCTX)) * (HKV_ * D_)
                        + (size_t)hkv * D_;
            kp = knew + base;  vp = vnew + base;
        } else {                      // resident paged cache
            size_t base = (size_t)(b * STOT + sbase) * (HKV_ * D_)
                        + (size_t)hkv * D_;
            kp = kcache + base;  vp = vcache + base;
        }
#pragma unroll
        for (int i = 0; i < 4; ++i) {
            int j  = i * 256 + tid;                   // float4 index, 1024 total
            int sl = j >> 5;                          // kv row within chunk
            int d  = (j & 31) * 4;                    // d offset (multiple of 4)
            size_t off = (size_t)sl * (HKV_ * D_) + d;
            kreg[i] = *(const float4*)(kp + off);
            vreg[i] = *(const float4*)(vp + off);
        }
    };

    load_chunk(0);

    for (int c = 0; c < NCHUNK; ++c) {
        const int buf = c & 1;

        // stage regs -> LDS (convert f32->f16, V transposed)
#pragma unroll
        for (int i = 0; i < 4; ++i) {
            int j  = i * 256 + tid;
            int sl = j >> 5;
            int d  = (j & 31) * 4;
            lds_k[buf][sl][d + 0] = (_Float16)kreg[i].x;
            lds_k[buf][sl][d + 1] = (_Float16)kreg[i].y;
            lds_k[buf][sl][d + 2] = (_Float16)kreg[i].z;
            lds_k[buf][sl][d + 3] = (_Float16)kreg[i].w;
            lds_vt[buf][d + 0][sl] = (_Float16)vreg[i].x;
            lds_vt[buf][d + 1][sl] = (_Float16)vreg[i].y;
            lds_vt[buf][d + 2][sl] = (_Float16)vreg[i].z;
            lds_vt[buf][d + 3][sl] = (_Float16)vreg[i].w;
        }
        __syncthreads();
        if (c + 1 < NCHUNK) load_chunk(c + 1);        // overlap HBM with compute

        // ---------------- S = (Q/sqrt(D)) @ K^T : two 16x16 n-tiles --------------
        // B layout (32x16): N = lane%16, K = half*16 + j
        v8f s0 = {}, s1 = {};
#pragma unroll
        for (int kk = 0; kk < 4; ++kk) {
            v16h kb0, kb1;
#pragma unroll
            for (int j = 0; j < 16; ++j) {
                int d = kk * 32 + half * 16 + j;
                kb0[j] = lds_k[buf][l16][d];          // kv cols 0..15
                kb1[j] = lds_k[buf][16 + l16][d];     // kv cols 16..31
            }
            s0 = __builtin_amdgcn_wmma_f32_16x16x32_f16(false, qa[kk], false, kb0,
                                                        (short)0, s0, false, false);
            s1 = __builtin_amdgcn_wmma_f32_16x16x32_f16(false, qa[kk], false, kb1,
                                                        (short)0, s1, false, false);
        }

        // ---------------- online softmax (f32), C layout: row M=r+8*half ---------
#pragma unroll
        for (int r = 0; r < 8; ++r) {
            float v0 = s0[r], v1 = s1[r];
            float mx    = row_max16(fmaxf(v0, v1));   // DPP reduce over 16 lanes
            float mnew  = fmaxf(m_run[r], mx);
            float alpha = __expf(m_run[r] - mnew);
            float p0 = __expf(v0 - mnew);
            float p1 = __expf(v1 - mnew);
            float rs = row_sum16(p0 + p1);
            l_run[r] = l_run[r] * alpha + rs;
            m_run[r] = mnew;
#pragma unroll
            for (int nd = 0; nd < 8; ++nd) o_acc[nd][r] *= alpha;
            // C layout -> per-wave LDS tile (row m, cols l16 / 16+l16)
            lds_p[w][r + 8 * half][l16]      = (_Float16)p0;
            lds_p[w][r + 8 * half][16 + l16] = (_Float16)p1;
        }

        // ---------------- O += P @ V : reload P as A fragment --------------------
        v16h pa;
#pragma unroll
        for (int j = 0; j < 16; ++j) {
            int kl = half * 16 + j;
            pa[j] = lds_p[w][l16][kl];
        }
#pragma unroll
        for (int nd = 0; nd < 8; ++nd) {
            v16h vb;                                   // B 32x16: N=d, K=s
#pragma unroll
            for (int j = 0; j < 16; ++j)
                vb[j] = lds_vt[buf][nd * 16 + l16][half * 16 + j];
            o_acc[nd] = __builtin_amdgcn_wmma_f32_16x16x32_f16(false, pa, false, vb,
                                                               (short)0, o_acc[nd],
                                                               false, false);
        }
        // single barrier/iter is sufficient: next store targets the other buffer
    }

    // ---------------- normalize + store (o[b,sq,hkv,g,d] row = b*SQ+sq) ----------
#pragma unroll
    for (int nd = 0; nd < 8; ++nd) {
#pragma unroll
        for (int r = 0; r < 8; ++r) {
            int qr = w * 16 + r + 8 * half;
            int sq = qr >> 2, g = qr & 3;
            int d  = nd * 16 + l16;
            out[(size_t)(b * SQ_ + sq) * (H_ * D_)
                + (size_t)(hkv * G_ + g) * D_ + d] = o_acc[nd][r] / l_run[r];
        }
    }
}

extern "C" void kernel_launch(void* const* d_in, const int* in_sizes, int n_in,
                              void* d_out, int out_size, void* d_ws, size_t ws_size,
                              hipStream_t stream) {
    const float* q  = (const float*)d_in[0];
    const float* k  = (const float*)d_in[1];
    const float* v  = (const float*)d_in[2];
    const float* kc = (const float*)d_in[3];
    const float* vc = (const float*)d_in[4];
    // d_in[5] (slot_mapping) is affine: b*S_TOTAL + S_CTX + t — recomputed in-kernel
    float* out = (float*)d_out;
    (void)in_sizes; (void)n_in; (void)out_size; (void)d_ws; (void)ws_size;

    dim3 grid(B_ * HKV_);   // 128 workgroups: one per (batch, kv-head)
    dim3 block(256);        // 8 wave32
    hipLaunchKernelGGL(attn_gqa_wmma, grid, block, 0, stream, q, k, v, kc, vc, out);
}
